// E_GCL_21268678050290
// MI455X (gfx1250) — compile-verified, matrix operand order checked by reading, not run on hardware
//
#include <hip/hip_runtime.h>
#include <hip/hip_bf16.h>
#include <math.h>

typedef __attribute__((ext_vector_type(16))) _Float16 v16h;
typedef __attribute__((ext_vector_type(8)))  _Float16 v8h;
typedef __attribute__((ext_vector_type(4)))  _Float16 v4h;
typedef __attribute__((ext_vector_type(8)))  float    v8f;

#define BB   2
#define NN   8192
#define EE   131072           // 2^17
#define FF   128
#define HH   256
#define NHH  4
#define DD   64
#define FEAT 268
#define FEATP 288             // 9 chunks of 32
#define XK   384              // H + F
#define EPSV 1e-8f
#define THRV 1e-6f

// ---- workspace layout (bytes) ----
#define OFF_AGG  0u
#define SZ_AGG   ((size_t)BB*NN*HH*4)            // 16 MB
#define OFF_W1P  (SZ_AGG)                        // 9*16 tiles
#define SZ_W1P   ((size_t)9*16*512*2)
#define OFF_W2P  (OFF_W1P + SZ_W1P)              // 4 heads * 2*4 tiles
#define SZ_W2P   ((size_t)4*8*512*2)
#define OFF_CW1P (OFF_W2P + SZ_W2P)              // 8*16 tiles
#define SZ_CW1P  ((size_t)8*16*512*2)
#define OFF_NW1P (OFF_CW1P + SZ_CW1P)            // 12*16 tiles
#define SZ_NW1P  ((size_t)12*16*512*2)
#define OFF_NW2P (OFF_NW1P + SZ_NW1P)            // 8*8 tiles
#define SZ_NW2P  ((size_t)8*8*512*2)

__device__ __forceinline__ float silu_f(float x) {
    return x / (1.0f + __expf(-x));
}

// WMMA 16-bit fragment element mapping (ISA 7.12.2):
// lane L, half j  ->  k_local, n_local (B matrix) / m (A matrix)
__device__ __forceinline__ void frag_kn(int lane, int half, int& kl, int& nl) {
    int g = lane >> 4; nl = lane & 15;
    int v = half >> 1, p = half & 1;
    kl = ((v >> 2) << 4) + (g << 3) + ((v & 3) << 1) + p;
}

// A fragment from row-major f16 LDS tile: per lane = two contiguous 8-half runs
__device__ __forceinline__ v16h load_a(const _Float16* rm, int ldk, int kbase) {
    int lane = threadIdx.x & 31;
    int m = lane & 15, g = lane >> 4;
    const _Float16* p0 = rm + m * ldk + kbase + g * 8;
    union { v16h v; v8h h[2]; } u;
    u.h[0] = *reinterpret_cast<const v8h*>(p0);
    u.h[1] = *reinterpret_cast<const v8h*>(p0 + 16);
    return u.v;
}

// B fragment from packed global weights (contiguous per lane)
__device__ __forceinline__ v16h load_b(const _Float16* packed, int tile) {
    int lane = threadIdx.x & 31;
    return *reinterpret_cast<const v16h*>(packed + (size_t)tile * 512 + lane * 16);
}

// ---------------- weight packing ----------------
__global__ void pack_rowmajor(const float* __restrict__ src, _Float16* __restrict__ dst,
                              int NT, int total, int Ksrc, int Nsrc) {
    int idx = blockIdx.x * blockDim.x + threadIdx.x;
    if (idx >= total) return;
    int tile = idx >> 9;
    int lane = (idx >> 4) & 31;
    int half = idx & 15;
    int kc = tile / NT, nt = tile % NT;
    int kl, nl; frag_kn(lane, half, kl, nl);
    int k = kc * 32 + kl, n = nt * 16 + nl;
    float v = (k < Ksrc && n < Nsrc) ? src[(size_t)k * Nsrc + n] : 0.0f;
    dst[idx] = (_Float16)v;
}

__global__ void pack_w1(const float* __restrict__ w1, _Float16* __restrict__ dst) {
    int idx = blockIdx.x * blockDim.x + threadIdx.x;
    if (idx >= 9 * 16 * 512) return;
    int tile = idx >> 9;
    int lane = (idx >> 4) & 31;
    int half = idx & 15;
    int kc = tile / 16, nt = tile % 16;
    int kl, nl; frag_kn(lane, half, kl, nl);
    int k = kc * 32 + kl, n = nt * 16 + nl;
    int head = n >> 6, d = n & 63;
    float v = (k < FEAT) ? w1[((size_t)head * FEAT + k) * DD + d] : 0.0f;
    dst[idx] = (_Float16)v;
}

// ---------------- init: zero agg, coord_out = coord ----------------
__global__ void init_kernel(float* __restrict__ agg, const float* __restrict__ coord,
                            float* __restrict__ coordOut) {
    size_t stride = (size_t)gridDim.x * blockDim.x;
    for (size_t i = (size_t)blockIdx.x * blockDim.x + threadIdx.x;
         i < (size_t)BB * NN * HH; i += stride) agg[i] = 0.0f;
    for (size_t i = (size_t)blockIdx.x * blockDim.x + threadIdx.x;
         i < (size_t)BB * NN * 3; i += stride) coordOut[i] = coord[i];
}

// ---------------- fused edge kernel: 1 wave = 16 edges ----------------
__global__ void __launch_bounds__(128) edge_kernel(
    const float* __restrict__ h, const float* __restrict__ coord,
    const int* __restrict__ eidx,
    const _Float16* __restrict__ W1p, const _Float16* __restrict__ W2p,
    const _Float16* __restrict__ CW1p,
    const float* __restrict__ eb1, const float* __restrict__ eb2,
    const float* __restrict__ lng, const float* __restrict__ lnb,
    const float* __restrict__ cb1, const float* __restrict__ cw2,
    float* __restrict__ agg, float* __restrict__ coordOut)
{
    __shared__ _Float16 s_feat[4][16 * FEATP];
    __shared__ int   s_row[4][16];
    __shared__ int   s_col[4][16];
    __shared__ float s_diff[4][16 * 3];

    const int wave = threadIdx.x >> 5;
    const int lane = threadIdx.x & 31;
    _Float16* feat = s_feat[wave];
    int* rowv = s_row[wave];
    int* colv = s_col[wave];
    float* diffv = s_diff[wave];

    const long tileg = (long)blockIdx.x * 4 + wave;     // 16384 tiles total
    const long e0 = tileg * 16;
    const int b = (int)(e0 >> 17);                      // EE = 2^17; tiles never straddle b
    const int el0 = (int)(e0 & (EE - 1));

    // ---- geometry + edge indices (lanes 0-15) ----
    if (lane < 16) {
        int el = el0 + lane;
        int r = eidx[el], c = eidx[EE + el];
        rowv[lane] = r; colv[lane] = c;
        const float* ci = coord + ((size_t)b * NN + r) * 3;
        const float* ck = coord + ((size_t)b * NN + c) * 3;
        float cix = ci[0], ciy = ci[1], ciz = ci[2];
        float ckx = ck[0], cky = ck[1], ckz = ck[2];
        float dx = cix - ckx, dy = ciy - cky, dz = ciz - ckz;
        float radial = dx * dx + dy * dy + dz * dz;
        float dist = sqrtf(radial);
        float dotv = cix * ckx + ciy * cky + ciz * ckz;
        float inva = 1.0f / (dist + EPSV);
        float ax = dx * inva, ay = dy * inva, az = dz * inva;
        float crx = ciy * ckz - ciz * cky;
        float cry = ciz * ckx - cix * ckz;
        float crz = cix * cky - ciy * ckx;
        float nb = sqrtf(crx * crx + cry * cry + crz * crz);
        float invb = 1.0f / (nb + EPSV);
        float bx = crx * invb, by = cry * invb, bz = crz * invb;
        float cx2 = ay * bz - az * by, cy2 = az * bx - ax * bz, cz2 = ax * by - ay * bx;
        float la = sqrtf(ax * ax + ay * ay + az * az);
        float lb = sqrtf(bx * bx + by * by + bz * bz);
        float lc = sqrtf(cx2 * cx2 + cy2 * cy2 + cz2 * cz2);
        bool bad = (la < THRV) || (lb < THRV) || (lc < THRV);
        float so3[9];
        if (bad) {
            so3[0] = 1; so3[1] = 0; so3[2] = 0;
            so3[3] = 0; so3[4] = 1; so3[5] = 0;
            so3[6] = 0; so3[7] = 0; so3[8] = 1;
        } else {
            so3[0] = ax;  so3[1] = bx;  so3[2] = cx2;
            so3[3] = ay;  so3[4] = by;  so3[5] = cy2;
            so3[6] = az;  so3[7] = bz;  so3[8] = cz2;
        }
        diffv[lane * 3 + 0] = dx; diffv[lane * 3 + 1] = dy; diffv[lane * 3 + 2] = dz;
        _Float16* fr = feat + lane * FEATP;
        fr[256] = (_Float16)radial; fr[257] = (_Float16)dist; fr[258] = (_Float16)dotv;
#pragma unroll
        for (int j = 0; j < 9; ++j) fr[259 + j] = (_Float16)so3[j];
#pragma unroll
        for (int k = 268; k < FEATP; ++k) fr[k] = (_Float16)0.0f;
    }

    // ---- stage h[row] | h[col] into LDS f16 (16 edges x 256) ----
    for (int i = 0; i < 32; ++i) {
        int q = i * 32 + lane;          // 1024 float4 per tile
        int m = q >> 6;                 // edge in tile
        int part = q & 63;              // 0..31 = h[row], 32..63 = h[col]
        int node = (part < 32) ? rowv[m] : colv[m];
        int off4 = (part & 31) * 4;
        float4 v = *reinterpret_cast<const float4*>(h + ((size_t)b * NN + node) * FF + off4);
        v4h hv = { (_Float16)v.x, (_Float16)v.y, (_Float16)v.z, (_Float16)v.w };
        *reinterpret_cast<v4h*>(feat + m * FEATP + ((part < 32) ? 0 : FF) + off4) = hv;
    }

    const int g = lane >> 4, nl = lane & 15;
    v8f acc[16];

    // ---- M1: feat(16x288) @ W1(288x256) ----
#pragma unroll
    for (int t = 0; t < 16; ++t)
#pragma unroll
        for (int j = 0; j < 8; ++j) acc[t][j] = 0.0f;
#pragma unroll
    for (int kc = 0; kc < 9; ++kc) {
        v16h a = load_a(feat, FEATP, kc * 32);
#pragma unroll
        for (int t = 0; t < 16; ++t) {
            v16h bf = load_b(W1p, kc * 16 + t);
            acc[t] = __builtin_amdgcn_wmma_f32_16x16x32_f16(false, a, false, bf,
                                                            (short)0, acc[t], false, false);
        }
    }

    // ---- hmid = silu(acc + edge_b1) -> LDS (16 x 256, reuse feat region) ----
#pragma unroll
    for (int t = 0; t < 16; ++t) {
        int n = t * 16 + nl;
        float bias = eb1[n];
#pragma unroll
        for (int r = 0; r < 8; ++r) {
            int m = r + 8 * g;
            feat[m * HH + n] = (_Float16)silu_f(acc[t][r] + bias);
        }
    }

    // ---- M2: per-head hmid(16x64) @ W2[h](64x64) ----
#pragma unroll
    for (int t = 0; t < 16; ++t)
#pragma unroll
        for (int j = 0; j < 8; ++j) acc[t][j] = 0.0f;
#pragma unroll
    for (int hd = 0; hd < 4; ++hd) {
#pragma unroll
        for (int kc = 0; kc < 2; ++kc) {
            v16h a = load_a(feat, HH, hd * 64 + kc * 32);
#pragma unroll
            for (int nt = 0; nt < 4; ++nt) {
                int t = hd * 4 + nt;
                v16h bf = load_b(W2p, hd * 8 + kc * 4 + nt);
                acc[t] = __builtin_amdgcn_wmma_f32_16x16x32_f16(false, a, false, bf,
                                                                (short)0, acc[t], false, false);
            }
        }
    }

    // ---- + edge_b2, then LayerNorm over n=0..255 (in C-fragment layout) ----
    float s[8], ss[8];
#pragma unroll
    for (int r = 0; r < 8; ++r) { s[r] = 0.0f; ss[r] = 0.0f; }
#pragma unroll
    for (int t = 0; t < 16; ++t) {
        int n = t * 16 + nl;
        float b2 = eb2[n];
#pragma unroll
        for (int r = 0; r < 8; ++r) {
            float v = acc[t][r] + b2;
            acc[t][r] = v; s[r] += v; ss[r] += v * v;
        }
    }
#pragma unroll
    for (int mask = 1; mask <= 8; mask <<= 1)
#pragma unroll
        for (int r = 0; r < 8; ++r) {
            s[r]  += __shfl_xor(s[r],  mask);
            ss[r] += __shfl_xor(ss[r], mask);
        }
    float mu[8], rstd[8];
#pragma unroll
    for (int r = 0; r < 8; ++r) {
        mu[r] = s[r] * (1.0f / HH);
        float var = ss[r] * (1.0f / HH) - mu[r] * mu[r];
        rstd[r] = rsqrtf(var + 1e-5f);
    }

    // ---- edge_feat: normalize, scatter-add agg, store f16 to LDS for M3 ----
#pragma unroll
    for (int t = 0; t < 16; ++t) {
        int n = t * 16 + nl;
        float gg = lng[n], bb = lnb[n];
#pragma unroll
        for (int r = 0; r < 8; ++r) {
            int m = r + 8 * g;
            float v = (acc[t][r] - mu[r]) * rstd[r] * gg + bb;
            acc[t][r] = v;
            feat[m * HH + n] = (_Float16)v;
            atomicAdd(agg + ((size_t)b * NN + rowv[m]) * HH + n, v);
        }
    }

    // ---- M3: edge_feat(16x256) @ coord_w1(256x256) ----
#pragma unroll
    for (int t = 0; t < 16; ++t)
#pragma unroll
        for (int j = 0; j < 8; ++j) acc[t][j] = 0.0f;
#pragma unroll
    for (int kc = 0; kc < 8; ++kc) {
        v16h a = load_a(feat, HH, kc * 32);
#pragma unroll
        for (int t = 0; t < 16; ++t) {
            v16h bf = load_b(CW1p, kc * 16 + t);
            acc[t] = __builtin_amdgcn_wmma_f32_16x16x32_f16(false, a, false, bf,
                                                            (short)0, acc[t], false, false);
        }
    }

    // ---- w = silu(. + coord_b1) . coord_w2 ; reduce ; coord atomics ----
    float pw[8];
#pragma unroll
    for (int r = 0; r < 8; ++r) pw[r] = 0.0f;
#pragma unroll
    for (int t = 0; t < 16; ++t) {
        int n = t * 16 + nl;
        float cb = cb1[n], cw = cw2[n];
#pragma unroll
        for (int r = 0; r < 8; ++r) pw[r] += silu_f(acc[t][r] + cb) * cw;
    }
#pragma unroll
    for (int mask = 1; mask <= 8; mask <<= 1)
#pragma unroll
        for (int r = 0; r < 8; ++r) pw[r] += __shfl_xor(pw[r], mask);

    if ((lane & 15) < 8) {
        int r = lane & 15;
        int m = g * 8 + r;
        float w = pw[r];
        size_t base = ((size_t)b * NN + rowv[m]) * 3;
        atomicAdd(coordOut + base + 0, diffv[m * 3 + 0] * w);
        atomicAdd(coordOut + base + 1, diffv[m * 3 + 1] * w);
        atomicAdd(coordOut + base + 2, diffv[m * 3 + 2] * w);
    }
}

// ---------------- node MLP kernel: 1 wave = 16 nodes ----------------
__global__ void __launch_bounds__(128) node_kernel(
    const float* __restrict__ h, const float* __restrict__ agg,
    const _Float16* __restrict__ NW1p, const _Float16* __restrict__ NW2p,
    const float* __restrict__ nb1, const float* __restrict__ nb2,
    float* __restrict__ hOut)
{
    __shared__ _Float16 s_x[4][16 * XK];

    const int wave = threadIdx.x >> 5;
    const int lane = threadIdx.x & 31;
    _Float16* x = s_x[wave];

    const int tile = blockIdx.x * 4 + wave;     // 1024 tiles
    const size_t g0 = (size_t)tile * 16;        // global node base (b*N+node)

    // stage x = [h(128) | agg(256)] as f16, 16 nodes
    for (int i = 0; i < 48; ++i) {
        int q = i * 32 + lane;                  // 1536 float4 per tile
        int m = q / 96;
        int part = q % 96;                      // 0..31 h, 32..95 agg
        const float* src = (part < 32) ? (h + (g0 + m) * FF + part * 4)
                                       : (agg + (g0 + m) * HH + (part - 32) * 4);
        float4 v = *reinterpret_cast<const float4*>(src);
        v4h hv = { (_Float16)v.x, (_Float16)v.y, (_Float16)v.z, (_Float16)v.w };
        int off = (part < 32) ? part * 4 : FF + (part - 32) * 4;
        *reinterpret_cast<v4h*>(x + m * XK + off) = hv;
    }

    const int g = lane >> 4, nl = lane & 15;
    v8f acc[16];
#pragma unroll
    for (int t = 0; t < 16; ++t)
#pragma unroll
        for (int j = 0; j < 8; ++j) acc[t][j] = 0.0f;

    // M1: x(16x384) @ node_w1(384x256)
#pragma unroll
    for (int kc = 0; kc < 12; ++kc) {
        v16h a = load_a(x, XK, kc * 32);
#pragma unroll
        for (int t = 0; t < 16; ++t) {
            v16h bf = load_b(NW1p, kc * 16 + t);
            acc[t] = __builtin_amdgcn_wmma_f32_16x16x32_f16(false, a, false, bf,
                                                            (short)0, acc[t], false, false);
        }
    }

    // u = silu(acc + node_b1) -> LDS (reuse x region, stride 256)
#pragma unroll
    for (int t = 0; t < 16; ++t) {
        int n = t * 16 + nl;
        float b1 = nb1[n];
#pragma unroll
        for (int r = 0; r < 8; ++r) {
            int m = r + 8 * g;
            x[m * HH + n] = (_Float16)silu_f(acc[t][r] + b1);
        }
    }

    // M2: u(16x256) @ node_w2(256x128)
    v8f acc2[8];
#pragma unroll
    for (int t = 0; t < 8; ++t)
#pragma unroll
        for (int j = 0; j < 8; ++j) acc2[t][j] = 0.0f;
#pragma unroll
    for (int kc = 0; kc < 8; ++kc) {
        v16h a = load_a(x, HH, kc * 32);
#pragma unroll
        for (int t = 0; t < 8; ++t) {
            v16h bf = load_b(NW2p, kc * 8 + t);
            acc2[t] = __builtin_amdgcn_wmma_f32_16x16x32_f16(false, a, false, bf,
                                                             (short)0, acc2[t], false, false);
        }
    }

    // residual: h_out = h + (out + node_b2)
#pragma unroll
    for (int t = 0; t < 8; ++t) {
        int n = t * 16 + nl;
        float b2 = nb2[n];
#pragma unroll
        for (int r = 0; r < 8; ++r) {
            int m = r + 8 * g;
            size_t o = (g0 + m) * FF + n;
            hOut[o] = h[o] + acc2[t][r] + b2;
        }
    }
}

extern "C" void kernel_launch(void* const* d_in, const int* in_sizes, int n_in,
                              void* d_out, int out_size, void* d_ws, size_t ws_size,
                              hipStream_t stream) {
    const float* h     = (const float*)d_in[0];
    const float* coord = (const float*)d_in[1];
    const int*   eidx  = (const int*)d_in[2];
    const float* ew1   = (const float*)d_in[3];
    const float* eb1   = (const float*)d_in[4];
    const float* ew2   = (const float*)d_in[5];
    const float* eb2   = (const float*)d_in[6];
    const float* lng   = (const float*)d_in[7];
    const float* lnb   = (const float*)d_in[8];
    const float* nw1   = (const float*)d_in[9];
    const float* nb1   = (const float*)d_in[10];
    const float* nw2   = (const float*)d_in[11];
    const float* nb2   = (const float*)d_in[12];
    const float* cw1   = (const float*)d_in[13];
    const float* cb1   = (const float*)d_in[14];
    const float* cw2   = (const float*)d_in[15];

    float* out      = (float*)d_out;
    float* hOut     = out;
    float* coordOut = out + (size_t)BB * NN * FF;

    char* ws = (char*)d_ws;
    float*     agg  = (float*)(ws + OFF_AGG);
    _Float16*  W1p  = (_Float16*)(ws + OFF_W1P);
    _Float16*  W2p  = (_Float16*)(ws + OFF_W2P);
    _Float16*  CW1p = (_Float16*)(ws + OFF_CW1P);
    _Float16*  NW1p = (_Float16*)(ws + OFF_NW1P);
    _Float16*  NW2p = (_Float16*)(ws + OFF_NW2P);

    // pack weights into WMMA B-fragment layout (f16)
    pack_w1<<<(9 * 16 * 512 + 255) / 256, 256, 0, stream>>>(ew1, W1p);
    for (int hd = 0; hd < 4; ++hd) {
        pack_rowmajor<<<(2 * 4 * 512 + 255) / 256, 256, 0, stream>>>(
            ew2 + (size_t)hd * DD * DD, W2p + (size_t)hd * 8 * 512, 4, 2 * 4 * 512, 64, 64);
    }
    pack_rowmajor<<<(8 * 16 * 512 + 255) / 256, 256, 0, stream>>>(cw1, CW1p, 16, 8 * 16 * 512, 256, 256);
    pack_rowmajor<<<(12 * 16 * 512 + 255) / 256, 256, 0, stream>>>(nw1, NW1p, 16, 12 * 16 * 512, 384, 256);
    pack_rowmajor<<<(8 * 8 * 512 + 255) / 256, 256, 0, stream>>>(nw2, NW2p, 8, 8 * 8 * 512, 256, 128);

    init_kernel<<<2048, 256, 0, stream>>>(agg, coord, coordOut);

    // 262144 edges / 16 per wave / 4 waves per block
    edge_kernel<<<4096, 128, 0, stream>>>(h, coord, eidx, W1p, W2p, CW1p,
                                          eb1, eb2, lng, lnb, cb1, cw2, agg, coordOut);

    // 16384 nodes / 16 per wave / 4 waves per block
    node_kernel<<<256, 128, 0, stream>>>(h, agg, NW1p, NW2p, nb1, nb2, hOut);
}
